// MambaEncoder_82145544503495
// MI455X (gfx1250) — matrix operand order repaired and checked by reading, not compiled, not run on hardware
//
#include <hip/hip_runtime.h>
#include <hip/hip_bf16.h>
#include <math.h>

#define LSEQ   2048
#define DMODEL 768
#define DINNER 1536
#define NSTATE 16
#define DTRANK 48
#define DTPAD  64            // dt GEMM K padded to multiple of 32
#define XPW    80            // dt_rank + 2*n_state
#define NCHUNK 16
#define CHUNK  (LSEQ / NCHUNK)

typedef __attribute__((ext_vector_type(8)))  __bf16 v8bf;
typedef __attribute__((ext_vector_type(16))) __bf16 v16bf;
typedef __attribute__((ext_vector_type(8)))  float  v8f;

static __device__ __forceinline__ __bf16 f2bf(float f) {
  unsigned u = __builtin_bit_cast(unsigned, f);
  u += 0x7FFFu + ((u >> 16) & 1u);      // round-to-nearest-even
  unsigned short h = (unsigned short)(u >> 16);
  return __builtin_bit_cast(__bf16, h);
}

static __device__ __forceinline__ float silu_(float x) {
  return x / (1.f + __expf(-x));
}

// ---------------------------------------------------------------- embedding
__global__ void k_embed(const int* __restrict__ ids, const float* __restrict__ emb,
                        float* __restrict__ x) {
  int idx = blockIdx.x * blockDim.x + threadIdx.x;
  if (idx >= LSEQ * DMODEL) return;
  int l = idx / DMODEL, d = idx - l * DMODEL;
  x[idx] = emb[(long)ids[l] * DMODEL + d];
}

// ---------------------------------------------------------------- f32 -> bf16 conversions
__global__ void k_cvt(const float* __restrict__ src, __bf16* __restrict__ dst, int n) {
  int i = blockIdx.x * blockDim.x + threadIdx.x;
  if (i < n) dst[i] = f2bf(src[i]);
}

// dst[drows x dld] = bf16(src[r*sld+c]) for r<srows && c<scols, else 0
__global__ void k_cvt_pad(const float* __restrict__ src, int sld, int srows, int scols,
                          __bf16* __restrict__ dst, int dld, int drows) {
  int i = blockIdx.x * blockDim.x + threadIdx.x;
  if (i >= drows * dld) return;
  int r = i / dld, c = i - r * dld;
  float v = (r < srows && c < scols) ? src[(size_t)r * sld + c] : 0.f;
  dst[i] = f2bf(v);
}

// ---------------------------------------------------------------- rmsnorm (bf16 out: feeds GEMM only)
__global__ void k_rmsnorm(const float* __restrict__ x, const float* __restrict__ w,
                          __bf16* __restrict__ out) {
  int l = blockIdx.x;
  __shared__ float red[256];
  float s = 0.f;
  for (int j = threadIdx.x; j < DMODEL; j += 256) {
    float v = x[l * DMODEL + j];
    s += v * v;
  }
  red[threadIdx.x] = s;
  __syncthreads();
  for (int off = 128; off > 0; off >>= 1) {
    if (threadIdx.x < off) red[threadIdx.x] += red[threadIdx.x + off];
    __syncthreads();
  }
  float scale = rsqrtf(red[0] / (float)DMODEL + 1e-5f);
  for (int j = threadIdx.x; j < DMODEL; j += 256)
    out[l * DMODEL + j] = f2bf(x[l * DMODEL + j] * scale * w[j]);
}

// ---------------------------------------------------------------- bf16 WMMA GEMM, 16x(16*NT) per wave
// Out[M,N] = A[M,K] @ B[K,N]  (+ epilogue). A,B bf16 row-major, Out f32.
// EPI 0: store   EPI 1: += residual Ex[m*ldo+n]   EPI 2: softplus(acc + bias Ex[n])
// Launch geometry must tile N exactly: (gridDim.y * blockDim.y) * 16 * NT == N.
// K multiple of 32; M multiple of 16. No branches -> EXEC all-1s at every WMMA.
template <int EPI, int NT>
__global__ void k_gemm(const __bf16* __restrict__ A, int lda,
                       const __bf16* __restrict__ B, int ldb,
                       float* __restrict__ Out, int ldo,
                       const float* __restrict__ Ex,
                       int M, int N, int K) {
  const int lane = threadIdx.x;        // 0..31
  const int g    = lane >> 4;
  const int lr   = lane & 15;
  const int m0   = blockIdx.x * 16;
  const int nbase = (blockIdx.y * blockDim.y + threadIdx.y) * (16 * NT);

  v8f acc[NT];
  #pragma unroll
  for (int t = 0; t < NT; ++t) acc[t] = (v8f){};

  // Strength-reduced operand pointers: pure pointer bumps in the hot loop.
  const __bf16* aptr = A + (size_t)(m0 + lr) * lda + (g << 3);  // A fragment base
  const __bf16* bptr = B + (size_t)(g << 3) * ldb + nbase + lr; // B fragment base (col lr)
  const size_t  bstep = (size_t)32 * ldb;

  for (int k0 = 0; k0 < K; k0 += 32) {
    // A fragment: e<8 -> k0+8g+e ; e>=8 -> k0+16+8g+(e-8). Two 16B loads.
    v8bf alo = *(const v8bf*)(aptr);
    v8bf ahi = *(const v8bf*)(aptr + 16);
    v16bf a = __builtin_shufflevector(alo, ahi, 0, 1, 2, 3, 4, 5, 6, 7,
                                      8, 9, 10, 11, 12, 13, 14, 15);
    #pragma unroll
    for (int t = 0; t < NT; ++t) {
      v16bf b;                                                  // e<8 -> k0+8g+e ; e>=8 -> k0+16+8g+(e-8)
      #pragma unroll
      for (int e = 0; e < 16; ++e)
        b[e] = bptr[(size_t)(e < 8 ? e : e + 8) * ldb + t * 16];
      acc[t] = __builtin_amdgcn_wmma_f32_16x16x32_bf16(
          /*neg_a=*/false, a, /*neg_b=*/false, b,
          /*c_mod=*/(short)0, acc[t], /*reuse_a=*/false, /*reuse_b=*/false);
    }
    aptr += 32;
    bptr += bstep;
  }

  #pragma unroll
  for (int t = 0; t < NT; ++t) {
    int n = nbase + t * 16 + lr;
    #pragma unroll
    for (int r = 0; r < 8; ++r) {
      int m = m0 + r + (g << 3);
      float v = acc[t][r];
      if (EPI == 1) v += Ex[(size_t)m * ldo + n];                           // residual
      if (EPI == 2) { v += Ex[n]; v = (v > 20.f) ? v : log1pf(__expf(v)); } // bias+softplus
      Out[(size_t)m * ldo + n] = v;
    }
  }
}

// ---------------------------------------------------------------- causal depthwise conv + SiLU
// writes f32 (scan input) and bf16 (GEMM input)
__global__ void k_conv_silu(const float* __restrict__ xz, const float* __restrict__ cw,
                            const float* __restrict__ cb, float* __restrict__ out,
                            __bf16* __restrict__ outb) {
  int idx = blockIdx.x * blockDim.x + threadIdx.x;
  if (idx >= LSEQ * DINNER) return;
  int l = idx / DINNER, d = idx - l * DINNER;
  float acc = cb[d];
  #pragma unroll
  for (int k = 0; k < 4; ++k) {
    int ll = l - 3 + k;
    if (ll >= 0) acc += cw[d * 4 + k] * xz[(size_t)ll * (2 * DINNER) + d];
  }
  float v = silu_(acc);
  out[idx]  = v;
  outb[idx] = f2bf(v);
}

// ---------------------------------------------------------------- chunked selective scan
__global__ void k_scanA(const float* __restrict__ u, const float* __restrict__ delta,
                        const float* __restrict__ xdbl, const float* __restrict__ A_log,
                        float* __restrict__ Sbuf, float* __restrict__ qbuf) {
  int c = blockIdx.x;
  int d = blockIdx.y * 256 + threadIdx.x;
  float A[NSTATE], h[NSTATE];
  #pragma unroll
  for (int n = 0; n < NSTATE; ++n) { A[n] = -__expf(A_log[d * NSTATE + n]); h[n] = 0.f; }
  float S = 0.f;
  __shared__ float sB[NSTATE];
  int l0 = c * CHUNK;
  for (int t = 0; t < CHUNK; ++t) {
    int l = l0 + t;
    if (threadIdx.x < NSTATE) sB[threadIdx.x] = xdbl[l * XPW + DTRANK + threadIdx.x];
    __syncthreads();
    float dt = delta[(size_t)l * DINNER + d];
    float ul = u[(size_t)l * DINNER + d];
    S += dt;
    #pragma unroll
    for (int n = 0; n < NSTATE; ++n)
      h[n] = __expf(dt * A[n]) * h[n] + dt * sB[n] * ul;
    __syncthreads();
  }
  Sbuf[c * DINNER + d] = S;
  #pragma unroll
  for (int n = 0; n < NSTATE; ++n)
    qbuf[(size_t)(c * DINNER + d) * NSTATE + n] = h[n];
}

__global__ void k_scanP(const float* __restrict__ A_log, const float* __restrict__ Sbuf,
                        const float* __restrict__ qbuf, float* __restrict__ hinit) {
  int d = blockIdx.x * 256 + threadIdx.x;
  float A[NSTATE], h[NSTATE];
  #pragma unroll
  for (int n = 0; n < NSTATE; ++n) { A[n] = -__expf(A_log[d * NSTATE + n]); h[n] = 0.f; }
  for (int c = 0; c < NCHUNK; ++c) {
    #pragma unroll
    for (int n = 0; n < NSTATE; ++n)
      hinit[(size_t)(c * DINNER + d) * NSTATE + n] = h[n];
    float S = Sbuf[c * DINNER + d];
    #pragma unroll
    for (int n = 0; n < NSTATE; ++n)
      h[n] = __expf(A[n] * S) * h[n] + qbuf[(size_t)(c * DINNER + d) * NSTATE + n];
  }
}

__global__ void k_scanB(const float* __restrict__ u, const float* __restrict__ delta,
                        const float* __restrict__ xdbl, const float* __restrict__ A_log,
                        const float* __restrict__ Dv, const float* __restrict__ xz,
                        const float* __restrict__ hinit, __bf16* __restrict__ yg) {
  int c = blockIdx.x;
  int d = blockIdx.y * 256 + threadIdx.x;
  float A[NSTATE], h[NSTATE];
  #pragma unroll
  for (int n = 0; n < NSTATE; ++n) {
    A[n] = -__expf(A_log[d * NSTATE + n]);
    h[n] = hinit[(size_t)(c * DINNER + d) * NSTATE + n];
  }
  float Dd = Dv[d];
  __shared__ float sBC[2 * NSTATE];
  int l0 = c * CHUNK;
  for (int t = 0; t < CHUNK; ++t) {
    int l = l0 + t;
    if (threadIdx.x < 2 * NSTATE) sBC[threadIdx.x] = xdbl[l * XPW + DTRANK + threadIdx.x];
    __syncthreads();
    float dt = delta[(size_t)l * DINNER + d];
    float ul = u[(size_t)l * DINNER + d];
    float y = 0.f;
    #pragma unroll
    for (int n = 0; n < NSTATE; ++n) {
      h[n] = __expf(dt * A[n]) * h[n] + dt * sBC[n] * ul;
      y += h[n] * sBC[NSTATE + n];
    }
    y += ul * Dd;
    float r = xz[(size_t)l * (2 * DINNER) + DINNER + d];
    yg[(size_t)l * DINNER + d] = f2bf(y * silu_(r));
    __syncthreads();
  }
}

// ---------------------------------------------------------------- driver
extern "C" void kernel_launch(void* const* d_in, const int* in_sizes, int n_in,
                              void* d_out, int out_size, void* d_ws, size_t ws_size,
                              hipStream_t stream) {
  (void)in_sizes; (void)n_in; (void)out_size; (void)ws_size;
  const int*   ids        = (const int*)  d_in[0];
  const float* emb        = (const float*)d_in[1];
  const float* norm_w     = (const float*)d_in[2];
  const float* in_proj_w  = (const float*)d_in[3];
  const float* conv_w     = (const float*)d_in[4];
  const float* conv_b     = (const float*)d_in[5];
  const float* x_proj_w   = (const float*)d_in[6];
  const float* dt_proj_w  = (const float*)d_in[7];
  const float* dt_proj_b  = (const float*)d_in[8];
  const float* A_log      = (const float*)d_in[9];
  const float* Dvec       = (const float*)d_in[10];
  const float* out_proj_w = (const float*)d_in[11];

  float* x = (float*)d_out;            // residual stream lives in d_out

  char* p = (char*)d_ws;
  auto carve = [&](size_t bytes) -> void* {
    void* r = (void*)p;
    p += (bytes + 255) & ~(size_t)255;
    return r;
  };
  // f32 scratch
  float* xz    = (float*)carve((size_t)LSEQ * 2 * DINNER * 4);
  float* xr    = (float*)carve((size_t)LSEQ * DINNER * 4);
  float* xdbl  = (float*)carve((size_t)LSEQ * XPW * 4);
  float* delta = (float*)carve((size_t)LSEQ * DINNER * 4);
  float* Sbuf  = (float*)carve((size_t)NCHUNK * DINNER * 4);
  float* qbuf  = (float*)carve((size_t)NCHUNK * DINNER * NSTATE * 4);
  float* hinit = (float*)carve((size_t)NCHUNK * DINNER * NSTATE * 4);
  // bf16 scratch (GEMM operands)
  __bf16* xnb  = (__bf16*)carve((size_t)LSEQ * DMODEL * 2);
  __bf16* xrb  = (__bf16*)carve((size_t)LSEQ * DINNER * 2);
  __bf16* dtA  = (__bf16*)carve((size_t)LSEQ * DTPAD * 2);
  __bf16* ygb  = (__bf16*)carve((size_t)LSEQ * DINNER * 2);
  __bf16* wIn  = (__bf16*)carve((size_t)DMODEL * 2 * DINNER * 2);
  __bf16* wXp  = (__bf16*)carve((size_t)DINNER * XPW * 2);
  __bf16* wDt  = (__bf16*)carve((size_t)DTPAD * DINNER * 2);
  __bf16* wOut = (__bf16*)carve((size_t)DINNER * DMODEL * 2);

  k_embed<<<(LSEQ * DMODEL + 255) / 256, 256, 0, stream>>>(ids, emb, x);

  for (int i = 0; i < 2; ++i) {
    const float* ipw = in_proj_w  + (size_t)i * DMODEL * 2 * DINNER;
    const float* xpw = x_proj_w   + (size_t)i * DINNER * XPW;
    const float* dpw = dt_proj_w  + (size_t)i * DTRANK * DINNER;
    const float* opw = out_proj_w + (size_t)i * DINNER * DMODEL;
    const float* Al  = A_log      + (size_t)i * DINNER * NSTATE;

    // weight conversions (amortized over 128 M-tiles of reuse each)
    k_cvt<<<(DMODEL * 2 * DINNER + 255) / 256, 256, 0, stream>>>(ipw, wIn, DMODEL * 2 * DINNER);
    k_cvt<<<(DINNER * XPW + 255) / 256, 256, 0, stream>>>(xpw, wXp, DINNER * XPW);
    k_cvt_pad<<<(DTPAD * DINNER + 255) / 256, 256, 0, stream>>>(
        dpw, DINNER, DTRANK, DINNER, wDt, DINNER, DTPAD);
    k_cvt<<<(DINNER * DMODEL + 255) / 256, 256, 0, stream>>>(opw, wOut, DINNER * DMODEL);

    k_rmsnorm<<<LSEQ, 256, 0, stream>>>(x, norm_w + i * DMODEL, xnb);

    { // xz = xn @ in_proj_w   (2048x768 @ 768x3072): 3072 = 12 blocks * 4 waves * 64
      dim3 grid(LSEQ / 16, 12), blk(32, 4);
      k_gemm<0, 4><<<grid, blk, 0, stream>>>(xnb, DMODEL, wIn, 2 * DINNER,
                                             xz, 2 * DINNER, nullptr,
                                             LSEQ, 2 * DINNER, DMODEL);
    }

    k_conv_silu<<<(LSEQ * DINNER + 255) / 256, 256, 0, stream>>>(
        xz, conv_w + (size_t)i * DINNER * 4, conv_b + i * DINNER, xr, xrb);

    { // x_dbl = xr @ x_proj_w   (2048x1536 @ 1536x80): 80 = 5 waves * 16, NT=1
      dim3 grid(LSEQ / 16, 1), blk(32, 5);
      k_gemm<0, 1><<<grid, blk, 0, stream>>>(xrb, DINNER, wXp, XPW,
                                             xdbl, XPW, nullptr,
                                             LSEQ, XPW, DINNER);
    }

    // pad dt activations (cols 0..47 of x_dbl) to K=64 in bf16
    k_cvt_pad<<<(LSEQ * DTPAD + 255) / 256, 256, 0, stream>>>(
        xdbl, XPW, LSEQ, DTRANK, dtA, DTPAD, LSEQ);

    { // delta = softplus(dt @ dt_proj_w + b)   (2048x64 @ 64x1536): 1536 = 6*4*64
      dim3 grid(LSEQ / 16, 6), blk(32, 4);
      k_gemm<2, 4><<<grid, blk, 0, stream>>>(dtA, DTPAD, wDt, DINNER,
                                             delta, DINNER, dt_proj_b + i * DINNER,
                                             LSEQ, DINNER, DTPAD);
    }

    // chunked selective scan (16 chunks of 128 -> 24K-way parallel)
    k_scanA<<<dim3(NCHUNK, DINNER / 256), 256, 0, stream>>>(xr, delta, xdbl, Al, Sbuf, qbuf);
    k_scanP<<<DINNER / 256, 256, 0, stream>>>(Al, Sbuf, qbuf, hinit);
    k_scanB<<<dim3(NCHUNK, DINNER / 256), 256, 0, stream>>>(xr, delta, xdbl, Al,
                                                            Dvec + i * DINNER, xz, hinit, ygb);

    { // x = x + yg @ out_proj_w   (2048x1536 @ 1536x768): 768 = 3*4*64
      dim3 grid(LSEQ / 16, 3), blk(32, 4);
      k_gemm<1, 4><<<grid, blk, 0, stream>>>(ygb, DINNER, wOut, DMODEL,
                                             x, DMODEL, x,
                                             LSEQ, DMODEL, DINNER);
    }
  }
}
